// ClassPixelMotifGraphRetrieval_3186865734150
// MI455X (gfx1250) — matrix operand order repaired and checked by reading, not compiled, not run on hardware
//
#include <hip/hip_runtime.h>
#include <hip/hip_bf16.h>
#include <cstddef>

// ---------------------------------------------------------------------------
// ClassPixelMotifGraphRetrieval — fully fused, single pass, WMMA-based sim.
//
// Shapes: B=16, N=9216, E=72580, C=7, H=64, ND=7, ED=5.
// Per item (node n or edge e), one wave computes:
//   A[16x64] = l2norm(mask * gelu(ln(x @ W + b)))       (f16, LDS staged)
//   Bt[7x64] = l2norm(proto[c, item, :])                (f16, LDS staged)
//   S[16x16] = A x Bt^T  via 2x v_wmma_f32_16x16x32_f16 (K = 64)
//   attn = sigmoid(S/0.2) * sigmoid(gate) (* mask) ; num += attn*S ; den += attn
// Block-reduce num/den in LDS (ds_add_f32), atomically fold to workspace,
// finalize kernel emits logits[16,7].
//
// Bandwidth-bound: streams edge_proto (130 MB) + node_proto (16.5 MB) +
// edge_attr (23 MB) + x (4 MB) once; ~176 MB @ 23.3 TB/s ≈ 8 us floor.
// ---------------------------------------------------------------------------

typedef _Float16 v16h __attribute__((ext_vector_type(16)));
typedef _Float16 v8h  __attribute__((ext_vector_type(8)));
typedef float    v8f  __attribute__((ext_vector_type(8)));

#define WAVES 8
#define BDIM  (WAVES * 32)

__device__ __forceinline__ float fast_rcp(float x) {
    return __builtin_amdgcn_rcpf(x);           // v_rcp_f32 (avoids div_scale chains)
}
__device__ __forceinline__ float sigf(float z) {
    return fast_rcp(1.0f + __expf(-z));
}

template<int D, bool HASMASK>
__global__ __launch_bounds__(BDIM) void fused_sim_kernel(
    const float* __restrict__ xin,      // [16, NI, D]
    const float* __restrict__ W,        // [D, 64]
    const float* __restrict__ bias,     // [64]
    const float* __restrict__ lng,      // [64]
    const float* __restrict__ lnb,      // [64]
    const float* __restrict__ proto,    // [7, NI, 64]
    const float* __restrict__ gateLog,  // [7, NI]
    const float* __restrict__ mask,     // [16, NI] (node only)
    float* __restrict__ gNum,           // [16*7]
    float* __restrict__ gDen,           // [16*7]
    int NI)
{
    __shared__ _Float16 Astage[WAVES][16][64];   // M=batch rows
    __shared__ _Float16 Bstage[WAVES][16][64];   // N=class rows (B^T), rows 7..15 zero
    __shared__ float sMask[WAVES][16];           // per-item mask row cache (node only)
    __shared__ float sW[7][64];
    __shared__ float sBias[64], sG[64], sBeta[64];
    __shared__ float sNum[16][16], sDen[16][16];

    const int tid  = threadIdx.x;
    const int lane = tid & 31;
    const int wave = tid >> 5;

    // ---- one-time block staging of tiny weights + zero reduction arrays ----
    for (int i = tid; i < D * 64; i += BDIM) sW[i >> 6][i & 63] = W[i];
    for (int i = tid; i < 64; i += BDIM) { sBias[i] = bias[i]; sG[i] = lng[i]; sBeta[i] = lnb[i]; }
    for (int i = tid; i < 256; i += BDIM) { (&sNum[0][0])[i] = 0.f; (&sDen[0][0])[i] = 0.f; }
    // zero unused B^T rows 7..15 of this wave's staging (written once, never touched again)
    for (int i = lane; i < 9 * 64; i += 32) Bstage[wave][7 + (i >> 6)][i & 63] = (_Float16)0.f;
    __syncthreads();

    const int   gw = blockIdx.x * WAVES + wave;
    const int   nw = gridDim.x * WAVES;
    const int   brow = lane >> 1;         // encoder: batch row handled by this lane
    const int   half = lane & 1;          // encoder: H-half (0..31 / 32..63)
    const int   h0   = half * 32;
    const int   ccol = lane & 15;         // WMMA D: class column of this lane

    float num[8], den[8];
#pragma unroll
    for (int j = 0; j < 8; ++j) { num[j] = 0.f; den[j] = 0.f; }

    for (int item = gw; item < NI; item += nw) {
        // ================= Phase 1: encoder -> Astage (all 32 lanes) ========
        float xi[D];
        {
            const float* xr = xin + ((size_t)brow * NI + item) * D;
#pragma unroll
            for (int d = 0; d < D; ++d) xi[d] = xr[d];
        }
        float y[32];
#pragma unroll
        for (int j = 0; j < 32; ++j) {
            const int h = h0 + j;
            float acc = sBias[h];
#pragma unroll
            for (int d = 0; d < D; ++d) acc = fmaf(xi[d], sW[d][h], acc);
            y[j] = acc;
        }
        // LayerNorm over H=64 (pairwise lane reduce via shfl_xor 1)
        float s = 0.f;
#pragma unroll
        for (int j = 0; j < 32; ++j) s += y[j];
        s += __shfl_xor(s, 1, 32);
        const float mean = s * (1.0f / 64.0f);
        float v = 0.f;
#pragma unroll
        for (int j = 0; j < 32; ++j) { const float t = y[j] - mean; v = fmaf(t, t, v); }
        v += __shfl_xor(v, 1, 32);
        const float rstd = rsqrtf(v * (1.0f / 64.0f) + 1e-5f);

        float maskv = 1.0f;
        if (HASMASK) {
            maskv = mask[(size_t)brow * NI + item];
            if (half == 0) sMask[wave][brow] = maskv;   // cache row for phase 4
        }

        float ss = 0.f;
#pragma unroll
        for (int j = 0; j < 32; ++j) {
            const int h = h0 + j;
            float t = fmaf((y[j] - mean) * rstd, sG[h], sBeta[h]);
            t = 0.5f * t * (1.0f + erff(t * 0.70710678118654752f));   // exact GELU
            if (HASMASK) t *= maskv;
            y[j] = t;
            ss = fmaf(t, t, ss);
        }
        ss += __shfl_xor(ss, 1, 32);
        const float inv = fast_rcp(fmaxf(sqrtf(ss), 1e-6f));
#pragma unroll
        for (int j = 0; j < 32; ++j)
            Astage[wave][brow][h0 + j] = (_Float16)(y[j] * inv);

        // ================= Phase 2: proto -> Bstage (lanes 0..27) ===========
        if (lane < 28) {
            const int c = lane >> 2, q = lane & 3;
            const float4* pr = reinterpret_cast<const float4*>(
                proto + ((size_t)c * NI + item) * 64 + q * 16);
            float p[16];
#pragma unroll
            for (int k = 0; k < 4; ++k) {
                const float4 f = pr[k];
                p[4 * k + 0] = f.x; p[4 * k + 1] = f.y; p[4 * k + 2] = f.z; p[4 * k + 3] = f.w;
            }
            float ps = 0.f;
#pragma unroll
            for (int k = 0; k < 16; ++k) ps = fmaf(p[k], p[k], ps);
            ps += __shfl_xor(ps, 1, 32);
            ps += __shfl_xor(ps, 2, 32);
            const float pinv = fast_rcp(fmaxf(sqrtf(ps), 1e-6f));
#pragma unroll
            for (int k = 0; k < 16; ++k)
                Bstage[wave][c][q * 16 + k] = (_Float16)(p[k] * pinv);
        }
        __builtin_amdgcn_wave_barrier();   // LDS writes (ph1/2) before reads (ph3/4)

        // ================= Phase 3: WMMA fragments + matmul =================
        // A (16x32 f16): lanes 0-15 -> M=lane, K 0..15;
        //                lanes 16-31 -> M=lane-16, K 16..31
        const int ka = (lane < 16) ? 0 : 16;
        const _Float16* Ar = &Astage[wave][ccol][0];
        v8h  al0 = *reinterpret_cast<const v8h*>(Ar + ka);
        v8h  ah0 = *reinterpret_cast<const v8h*>(Ar + ka + 8);
        v8h  al1 = *reinterpret_cast<const v8h*>(Ar + ka + 32);
        v8h  ah1 = *reinterpret_cast<const v8h*>(Ar + ka + 40);
        v16h a0 = __builtin_shufflevector(al0, ah0, 0,1,2,3,4,5,6,7,8,9,10,11,12,13,14,15);
        v16h a1 = __builtin_shufflevector(al1, ah1, 0,1,2,3,4,5,6,7,8,9,10,11,12,13,14,15);
        // B (32x16 f16), column per lane: lanes 0-15 K 0..15, lanes 16-31 K 16..31
        const int kb = (lane < 16) ? 0 : 16;
        const _Float16* Br = &Bstage[wave][ccol][0];
        v16h b0 = *reinterpret_cast<const v16h*>(Br + kb);
        v16h b1 = *reinterpret_cast<const v16h*>(Br + kb + 32);

        v8f acc = {};
        acc = __builtin_amdgcn_wmma_f32_16x16x32_f16(false, a0, false, b0,
                                                     (short)0, acc, false, false);
        acc = __builtin_amdgcn_wmma_f32_16x16x32_f16(false, a1, false, b1,
                                                     (short)0, acc, false, false);

        // ================= Phase 4: gate/sigmoid + accumulate ===============
        if (ccol < 7) {
            const float gate = sigf(gateLog[(size_t)ccol * NI + item]);
#pragma unroll
            for (int j = 0; j < 8; ++j) {
                const int bb = (lane < 16) ? j : (j + 8);   // D layout: M=j or j+8
                const float sim = acc[j];
                float attn = gate * sigf(sim * 5.0f);       // 1/TEMP = 5
                if (HASMASK) attn *= sMask[wave][bb];       // LDS-cached mask row
                num[j] = fmaf(attn, sim, num[j]);
                den[j] += attn;
            }
        }
        __builtin_amdgcn_wave_barrier();   // keep next iter's LDS writes behind reads
    }

    // ================= block reduction =================
    if (ccol < 7) {
#pragma unroll
        for (int j = 0; j < 8; ++j) {
            const int bb = (lane < 16) ? j : (j + 8);
            atomicAdd(&sNum[bb][ccol], num[j]);
            atomicAdd(&sDen[bb][ccol], den[j]);
        }
    }
    __syncthreads();
    if (tid < 112) {
        const int bb = tid / 7, cc = tid % 7;
        atomicAdd(&gNum[bb * 7 + cc], sNum[bb][cc]);
        atomicAdd(&gDen[bb * 7 + cc], sDen[bb][cc]);
    }
}

__global__ void zero_ws_kernel(float* ws) {
    const int i = blockIdx.x * blockDim.x + threadIdx.x;
    if (i < 448) ws[i] = 0.f;
}

__global__ void finalize_kernel(const float* __restrict__ ws, float* __restrict__ out) {
    const int i = threadIdx.x;
    if (i < 112) {
        const float nn = ws[i],       dn = ws[112 + i];
        const float ne = ws[224 + i], de = ws[336 + i];
        out[i] = nn / fmaxf(dn, 1e-6f) + 0.5f * (ne / fmaxf(de, 1e-6f));
    }
}

extern "C" void kernel_launch(void* const* d_in, const int* in_sizes, int n_in,
                              void* d_out, int out_size, void* d_ws, size_t ws_size,
                              hipStream_t stream) {
    (void)in_sizes; (void)n_in; (void)out_size; (void)ws_size;
    const int N = 9216, E = 72580;

    const float* x          = (const float*)d_in[0];
    const float* edge_attr  = (const float*)d_in[1];
    // d_in[2] = edge_index (unused: gnn_layers == 0)
    const float* node_mask  = (const float*)d_in[3];
    const float* Wn         = (const float*)d_in[4];
    const float* bn         = (const float*)d_in[5];
    const float* gn_g       = (const float*)d_in[6];
    const float* gn_b       = (const float*)d_in[7];
    const float* We         = (const float*)d_in[8];
    const float* be         = (const float*)d_in[9];
    const float* ge_g       = (const float*)d_in[10];
    const float* ge_b       = (const float*)d_in[11];
    const float* node_proto = (const float*)d_in[12];
    const float* edge_proto = (const float*)d_in[13];
    const float* node_gate  = (const float*)d_in[14];
    const float* edge_gate  = (const float*)d_in[15];

    float* ws  = (float*)d_ws;
    float* out = (float*)d_out;

    zero_ws_kernel<<<1, 512, 0, stream>>>(ws);

    fused_sim_kernel<7, true><<<128, BDIM, 0, stream>>>(
        x, Wn, bn, gn_g, gn_b, node_proto, node_gate, node_mask,
        ws + 0, ws + 112, N);

    fused_sim_kernel<5, false><<<512, BDIM, 0, stream>>>(
        edge_attr, We, be, ge_g, ge_b, edge_proto, edge_gate, nullptr,
        ws + 224, ws + 336, E);

    finalize_kernel<<<1, 128, 0, stream>>>(ws, out);
}
